// MultiHeadAttention_43508018709076
// MI455X (gfx1250) — matrix operand order repaired and checked by reading, not compiled, not run on hardware
//
#include <hip/hip_runtime.h>
#include <hip/hip_bf16.h>

typedef __attribute__((ext_vector_type(16))) __bf16 bf16x16;
typedef __attribute__((ext_vector_type(8)))  __bf16 bf16x8;
typedef __attribute__((ext_vector_type(8)))  float  v8f;
typedef __attribute__((ext_vector_type(4)))  unsigned int v4u;
typedef __attribute__((ext_vector_type(8)))  int v8i;
typedef __attribute__((ext_vector_type(4)))  int v4i;

#define LOG2E 1.4426950408889634f
#define LDP 56   // LDS row stride (bf16): 112B = 64B tile row + 48B TDM pad
#define PSTR 40  // P-scratch row stride: 80B, 16B-aligned

static __device__ __forceinline__ bf16x16 ld16(const __bf16* p) {
  bf16x8 a = *(const bf16x8*)p;
  bf16x8 b = *(const bf16x8*)(p + 8);
  bf16x16 r;
#pragma unroll
  for (int i = 0; i < 8; ++i) { r[i] = a[i]; r[i + 8] = b[i]; }
  return r;
}
static __device__ __forceinline__ bf16x16 ld8_8(const __bf16* p0, const __bf16* p1) {
  bf16x8 a = *(const bf16x8*)p0;
  bf16x8 b = *(const bf16x8*)p1;
  bf16x16 r;
#pragma unroll
  for (int i = 0; i < 8; ++i) { r[i] = a[i]; r[i + 8] = b[i]; }
  return r;
}
static __device__ __forceinline__ v8f wmma_bf16(bf16x16 a, bf16x16 b, v8f c) {
  return __builtin_amdgcn_wmma_f32_16x16x32_bf16(false, a, false, b, (short)0, c,
                                                 false, false);
}

// ---------------------------------------------------------------------------
// TDM: async 2-D tile load (bf16) global -> LDS with row padding.
// Tile is tile_rows x 32 bf16; each 64B row is followed by 48B of LDS pad
// (pad_interval code 3 = 16 DWORDs, pad_amount code 11 = 12 DWORDs), giving
// the 112B (= LDP) conflict-free row stride the compute fragments expect.
// ---------------------------------------------------------------------------
static __device__ __forceinline__ void tdm_load_tile32(
    unsigned int lds_addr, const __bf16* gptr, unsigned int tensor_rows,
    unsigned int row_len, unsigned int tile_rows) {
  const unsigned long long ga = (unsigned long long)(uintptr_t)gptr;
  v4u g0;
  g0[0] = 1u;                                   // count=1 (valid), user mode
  g0[1] = lds_addr;                             // LDS byte offset
  g0[2] = (unsigned int)ga;                     // global addr [31:0]
  g0[3] = (unsigned int)(ga >> 32) | (2u << 30);// global addr [56:32] | type=2
  v8i g1;
  g1[0] = (int)((1u << 16)   // data_size = 2B
              | (1u << 20)   // pad_enable
              | (3u << 22)   // pad_interval: 16 DWORDs
              | (11u << 25));// pad_amount: 12 DWORDs
  g1[1] = (int)(row_len << 16);                         // tensor_dim0 lo16
  g1[2] = (int)((row_len >> 16) | (tensor_rows << 16)); // dim0 hi / dim1 lo
  g1[3] = (int)((tensor_rows >> 16) | (32u << 16));     // dim1 hi / tile_dim0=32
  g1[4] = (int)tile_rows;                               // tile_dim1 (tile_dim2=0)
  g1[5] = (int)row_len;                                 // tensor_dim0_stride lo32
  g1[6] = 0;
  g1[7] = 0;
  v4i z = {0, 0, 0, 0};
#if defined(__clang_major__) && (__clang_major__ >= 23)
  v8i z8 = {0, 0, 0, 0, 0, 0, 0, 0};
  __builtin_amdgcn_tensor_load_to_lds(g0, g1, z, z, z8, 0);
#else
  __builtin_amdgcn_tensor_load_to_lds(g0, g1, z, z, 0);
#endif
}

// ---------------------------------------------------------------------------
// Kernel 0: fp32 -> bf16 convert (feeds the TDM-staged GEMMs).
// ---------------------------------------------------------------------------
__global__ void __launch_bounds__(256) cvt_bf16_kernel(
    const float* __restrict__ in, __bf16* __restrict__ out, int n8) {
  const int id = blockIdx.x * 256 + threadIdx.x;
  if (id >= n8) return;
  const float4* s = (const float4*)(in + (size_t)id * 8);
  const float4 f0 = s[0], f1 = s[1];
  bf16x8 v;
  v[0] = (__bf16)f0.x; v[1] = (__bf16)f0.y; v[2] = (__bf16)f0.z; v[3] = (__bf16)f0.w;
  v[4] = (__bf16)f1.x; v[5] = (__bf16)f1.y; v[6] = (__bf16)f1.z; v[7] = (__bf16)f1.w;
  *(bf16x8*)(out + (size_t)id * 8) = v;
}

// ---------------------------------------------------------------------------
// Kernel 1: fused QKV projection, TDM double-buffered.
// Block = 256 thr (8 waves), tile M=128 x N=64, K-step 32, 12 WMMA/step.
// Q,K -> [B,H,L,D] bf16 ; V -> [B,H,D,L] bf16 (transposed for P@V).
// ---------------------------------------------------------------------------
__global__ void __launch_bounds__(256) qkv_proj_kernel(
    const __bf16* __restrict__ xb, const __bf16* __restrict__ wqb,
    const __bf16* __restrict__ wkb, const __bf16* __restrict__ wvb,
    __bf16* __restrict__ qh, __bf16* __restrict__ kh, __bf16* __restrict__ vt) {
  __shared__ __align__(16) __bf16 xs[2 * 128 * LDP];
  __shared__ __align__(16) __bf16 ws[2 * 3 * 64 * LDP];

  const int tid = threadIdx.x;
  const int lane = tid & 31, wave = tid >> 5;
  const int half = lane >> 4, nn = lane & 15;
  const int m0 = blockIdx.x * 128;  // rows over B*L = 4096
  const int n0 = blockIdx.y * 64;   // cols over C = 1024

  const unsigned int xs_lds = (unsigned int)(uintptr_t)xs;
  const unsigned int ws_lds = (unsigned int)(uintptr_t)ws;
  const unsigned int XSB = 128 * LDP * 2;      // bytes per x buffer
  const unsigned int WSB = 3 * 64 * LDP * 2;   // bytes per W buffer
  const unsigned int WOFF = 64 * LDP * 2;      // bytes per single W tile

  v8f acc[3][4];
#pragma unroll
  for (int w = 0; w < 3; ++w)
#pragma unroll
    for (int t = 0; t < 4; ++t) acc[w][t] = {};

  if (wave == 0) {  // prologue: tile 0 into buffer 0
    tdm_load_tile32(xs_lds, xb + (size_t)m0 * 1024, 4096, 1024, 128);
    tdm_load_tile32(ws_lds, wqb + (size_t)n0 * 1024, 1024, 1024, 64);
    tdm_load_tile32(ws_lds + WOFF, wkb + (size_t)n0 * 1024, 1024, 1024, 64);
    tdm_load_tile32(ws_lds + 2 * WOFF, wvb + (size_t)n0 * 1024, 1024, 1024, 64);
  }

  for (int it = 0; it < 32; ++it) {
    __syncthreads();  // all waves done reading the buffer TDM is about to fill
    if (wave == 0) {
      if (it + 1 < 32) {
        const int kn = (it + 1) * 32;
        const unsigned int nb = (unsigned int)((it + 1) & 1);
        tdm_load_tile32(xs_lds + nb * XSB, xb + (size_t)m0 * 1024 + kn, 4096, 1024, 128);
        tdm_load_tile32(ws_lds + nb * WSB, wqb + (size_t)n0 * 1024 + kn, 1024, 1024, 64);
        tdm_load_tile32(ws_lds + nb * WSB + WOFF, wkb + (size_t)n0 * 1024 + kn, 1024, 1024, 64);
        tdm_load_tile32(ws_lds + nb * WSB + 2 * WOFF, wvb + (size_t)n0 * 1024 + kn, 1024, 1024, 64);
        __builtin_amdgcn_s_wait_tensorcnt((short)4);  // current 4 tiles done
      } else {
        __builtin_amdgcn_s_wait_tensorcnt((short)0);
      }
    }
    __syncthreads();  // current buffer visible to all waves

    const int cur = it & 1;
    const __bf16* xcur = xs + cur * (128 * LDP);
    const __bf16* wcur = ws + cur * (3 * 64 * LDP);
    const __bf16* arow = xcur + (wave * 16 + nn) * LDP;
    const bf16x16 a = ld16(arow + 16 * half);
#pragma unroll
    for (int w = 0; w < 3; ++w) {
      bf16x16 bfr[4];
#pragma unroll
      for (int t = 0; t < 4; ++t)
        bfr[t] = ld16(wcur + (w * 64 + t * 16 + nn) * LDP + 16 * half);
#pragma unroll
      for (int t = 0; t < 4; ++t) acc[w][t] = wmma_bf16(a, bfr[t], acc[w][t]);
    }
  }

  const int b = m0 >> 11;
  const int lbase = (m0 & 2047) + wave * 16 + 8 * half;
#pragma unroll
  for (int t = 0; t < 4; ++t) {
    const int col = n0 + t * 16 + nn;
    const int h = col >> 6, d = col & 63;
#pragma unroll
    for (int r = 0; r < 8; ++r) {
      const int l = lbase + r;
      const size_t iqk = ((size_t)(b * 16 + h) * 2048 + l) * 64 + d;
      qh[iqk] = (__bf16)acc[0][t][r];
      kh[iqk] = (__bf16)acc[1][t][r];
      const size_t iv = ((size_t)(b * 16 + h) * 64 + d) * 2048 + l;
      vt[iv] = (__bf16)acc[2][t][r];
    }
  }
}

// ---------------------------------------------------------------------------
// Kernel 2: flash attention with symmetric ALiBi, one wave = 16 queries x D=64.
// j-blocks of 32: 4 WMMA for S, online softmax (log2 domain), 4 WMMA for P@V.
// ---------------------------------------------------------------------------
__global__ void __launch_bounds__(128) attn_kernel(
    const __bf16* __restrict__ qh, const __bf16* __restrict__ kh,
    const __bf16* __restrict__ vt, __bf16* __restrict__ ao) {
  __shared__ __align__(16) __bf16 pbuf[4 * 16 * PSTR];

  const int lane = threadIdx.x & 31, wave = threadIdx.x >> 5;
  const int half = lane >> 4, nn = lane & 15;
  const int g = blockIdx.x * 4 + wave;  // 0..4095
  const int bh = g >> 7, qb = g & 127;
  const int i0 = qb * 16;
  const int b = bh >> 4, h = bh & 15;

  const float slope_l2 = exp2f(-0.5f * (float)(h + 1)) * LOG2E;
  const float kscale = 0.125f * LOG2E;  // 1/sqrt(64) * log2e

  const __bf16* Q = qh + (size_t)bh * 2048 * 64;
  const __bf16* K = kh + (size_t)bh * 2048 * 64;
  const __bf16* V = vt + (size_t)bh * 64 * 2048;
  __bf16* pb = pbuf + wave * 16 * PSTR;

  const __bf16* qrow = Q + (size_t)(i0 + nn) * 64;
  const bf16x16 a0 = ld16(qrow + 16 * half);
  const bf16x16 a1 = ld16(qrow + 32 + 16 * half);

  float mrow[8], lrow[8], irf[8];
  v8f O[4];
#pragma unroll
  for (int r = 0; r < 8; ++r) {
    mrow[r] = -1e30f;
    lrow[r] = 0.f;
    irf[r] = (float)(i0 + r + 8 * half);
  }
#pragma unroll
  for (int t = 0; t < 4; ++t) O[t] = {};

  for (int jb = 0; jb < 64; ++jb) {
    const int j0 = jb * 32;
    v8f s0 = {}, s1 = {};
    {
      const __bf16* k0p = K + (size_t)(j0 + nn) * 64 + 16 * half;
      const __bf16* k1p = K + (size_t)(j0 + 16 + nn) * 64 + 16 * half;
      bf16x16 kf[4];
      kf[0] = ld16(k0p);
      kf[1] = ld16(k0p + 32);
      kf[2] = ld16(k1p);
      kf[3] = ld16(k1p + 32);
      s0 = wmma_bf16(a0, kf[0], s0);
      s0 = wmma_bf16(a1, kf[1], s0);
      s1 = wmma_bf16(a0, kf[2], s1);
      s1 = wmma_bf16(a1, kf[3], s1);
    }
    const float c0 = (float)(j0 + nn), c1 = c0 + 16.f;
    float bmax[8];
#pragma unroll
    for (int r = 0; r < 8; ++r) {
      s0[r] = s0[r] * kscale - slope_l2 * fabsf(c0 - irf[r]);
      s1[r] = s1[r] * kscale - slope_l2 * fabsf(c1 - irf[r]);
      bmax[r] = fmaxf(s0[r], s1[r]);
    }
#pragma unroll
    for (int m = 1; m <= 8; m <<= 1)
#pragma unroll
      for (int r = 0; r < 8; ++r)
        bmax[r] = fmaxf(bmax[r], __shfl_xor(bmax[r], m, 32));
    float fac[8], rs[8];
#pragma unroll
    for (int r = 0; r < 8; ++r) {
      const float mn = fmaxf(mrow[r], bmax[r]);
      const float p0 = exp2f(s0[r] - mn);
      const float p1 = exp2f(s1[r] - mn);
      fac[r] = exp2f(mrow[r] - mn);
      mrow[r] = mn;
      s0[r] = p0;
      s1[r] = p1;
      rs[r] = p0 + p1;
    }
#pragma unroll
    for (int m = 1; m <= 8; m <<= 1)
#pragma unroll
      for (int r = 0; r < 8; ++r) rs[r] += __shfl_xor(rs[r], m, 32);
#pragma unroll
    for (int r = 0; r < 8; ++r) lrow[r] = lrow[r] * fac[r] + rs[r];
#pragma unroll
    for (int t = 0; t < 4; ++t)
#pragma unroll
      for (int r = 0; r < 8; ++r) O[t][r] *= fac[r];
    // P (C layout) -> LDS bf16 -> A-fragment layout
#pragma unroll
    for (int r = 0; r < 8; ++r) {
      __bf16* row = pb + (r + 8 * half) * PSTR;
      row[nn] = (__bf16)s0[r];
      row[16 + nn] = (__bf16)s1[r];
    }
    const __bf16* prow = pb + nn * PSTR;
    const bf16x16 pa = ld16(prow + 16 * half);
    bf16x16 vf[4];
#pragma unroll
    for (int t = 0; t < 4; ++t)
      vf[t] = ld16(V + (size_t)(t * 16 + nn) * 2048 + j0 + 16 * half);
#pragma unroll
    for (int t = 0; t < 4; ++t) O[t] = wmma_bf16(pa, vf[t], O[t]);
  }
#pragma unroll
  for (int r = 0; r < 8; ++r) lrow[r] = 1.0f / lrow[r];
  const size_t rowbase = (size_t)(b * 2048 + i0 + 8 * half);
#pragma unroll
  for (int t = 0; t < 4; ++t) {
    const int col = h * 64 + t * 16 + nn;
#pragma unroll
    for (int r = 0; r < 8; ++r)
      ao[(rowbase + r) * 1024 + col] = (__bf16)(O[t][r] * lrow[r]);
  }
}

// ---------------------------------------------------------------------------
// Kernel 3: output projection  out = ao @ Wo^T + bo  (fp32), TDM staged.
// ---------------------------------------------------------------------------
__global__ void __launch_bounds__(256) out_proj_kernel(
    const __bf16* __restrict__ ao, const __bf16* __restrict__ wob,
    const float* __restrict__ bo, float* __restrict__ out) {
  __shared__ __align__(16) __bf16 as_[2 * 128 * LDP];
  __shared__ __align__(16) __bf16 ws_[2 * 64 * LDP];

  const int tid = threadIdx.x;
  const int lane = tid & 31, wave = tid >> 5;
  const int half = lane >> 4, nn = lane & 15;
  const int m0 = blockIdx.x * 128, n0 = blockIdx.y * 64;

  const unsigned int as_lds = (unsigned int)(uintptr_t)as_;
  const unsigned int ws_lds = (unsigned int)(uintptr_t)ws_;
  const unsigned int ASB = 128 * LDP * 2;
  const unsigned int WSB = 64 * LDP * 2;

  v8f acc[4];
#pragma unroll
  for (int t = 0; t < 4; ++t) acc[t] = {};

  if (wave == 0) {
    tdm_load_tile32(as_lds, ao + (size_t)m0 * 1024, 4096, 1024, 128);
    tdm_load_tile32(ws_lds, wob + (size_t)n0 * 1024, 1024, 1024, 64);
  }

  for (int it = 0; it < 32; ++it) {
    __syncthreads();
    if (wave == 0) {
      if (it + 1 < 32) {
        const int kn = (it + 1) * 32;
        const unsigned int nb = (unsigned int)((it + 1) & 1);
        tdm_load_tile32(as_lds + nb * ASB, ao + (size_t)m0 * 1024 + kn, 4096, 1024, 128);
        tdm_load_tile32(ws_lds + nb * WSB, wob + (size_t)n0 * 1024 + kn, 1024, 1024, 64);
        __builtin_amdgcn_s_wait_tensorcnt((short)2);
      } else {
        __builtin_amdgcn_s_wait_tensorcnt((short)0);
      }
    }
    __syncthreads();

    const int cur = it & 1;
    const __bf16* arow = as_ + cur * (128 * LDP) + (wave * 16 + nn) * LDP;
    const __bf16* wcur = ws_ + cur * (64 * LDP);
    const bf16x16 a = ld16(arow + 16 * half);
    bf16x16 bfr[4];
#pragma unroll
    for (int t = 0; t < 4; ++t)
      bfr[t] = ld16(wcur + (t * 16 + nn) * LDP + 16 * half);
#pragma unroll
    for (int t = 0; t < 4; ++t) acc[t] = wmma_bf16(a, bfr[t], acc[t]);
  }
#pragma unroll
  for (int t = 0; t < 4; ++t) {
    const int col = n0 + t * 16 + nn;
    const float bias = bo[col];
    const int rbase = m0 + wave * 16 + 8 * half;
#pragma unroll
    for (int r = 0; r < 8; ++r)
      out[(size_t)(rbase + r) * 1024 + col] = acc[t][r] + bias;
  }
}

extern "C" void kernel_launch(void* const* d_in, const int* in_sizes, int n_in,
                              void* d_out, int out_size, void* d_ws, size_t ws_size,
                              hipStream_t stream) {
  (void)in_sizes; (void)n_in; (void)out_size; (void)ws_size;
  const float* q  = (const float*)d_in[0];
  const float* Wq = (const float*)d_in[1];
  const float* Wk = (const float*)d_in[2];
  const float* Wv = (const float*)d_in[3];
  const float* Wo = (const float*)d_in[4];
  const float* bo = (const float*)d_in[5];
  float* out = (float*)d_out;

  // workspace carve-up (bf16 elements)
  const size_t NQ = (size_t)4096 * 1024;  // 4 M
  const size_t NW = (size_t)1024 * 1024;  // 1 M
  __bf16* qh  = (__bf16*)d_ws;
  __bf16* kh  = qh + NQ;
  __bf16* vt  = kh + NQ;
  __bf16* ao  = vt + NQ;
  __bf16* xb  = ao + NQ;
  __bf16* wqb = xb + NQ;
  __bf16* wkb = wqb + NW;
  __bf16* wvb = wkb + NW;
  __bf16* wob = wvb + NW;

  cvt_bf16_kernel<<<dim3(2048), 256, 0, stream>>>(q, xb, (int)(NQ / 8));
  cvt_bf16_kernel<<<dim3(512), 256, 0, stream>>>(Wq, wqb, (int)(NW / 8));
  cvt_bf16_kernel<<<dim3(512), 256, 0, stream>>>(Wk, wkb, (int)(NW / 8));
  cvt_bf16_kernel<<<dim3(512), 256, 0, stream>>>(Wv, wvb, (int)(NW / 8));
  cvt_bf16_kernel<<<dim3(512), 256, 0, stream>>>(Wo, wob, (int)(NW / 8));

  qkv_proj_kernel<<<dim3(32, 16), 256, 0, stream>>>(xb, wqb, wkb, wvb, qh, kh, vt);
  attn_kernel<<<dim3(1024), 128, 0, stream>>>(qh, kh, vt, ao);
  out_proj_kernel<<<dim3(32, 16), 256, 0, stream>>>(ao, wob, bo, out);
}
